// GravityAttention_25615184954029
// MI455X (gfx1250) — matrix operand order repaired and checked by reading, not compiled, not run on hardware
//
#include <hip/hip_runtime.h>
#include <hip/hip_fp16.h>

// ---------------------------------------------------------------------------
// GravityAttention for MI455X (gfx1250, wave32, WMMA + async-LDS)
// B=2 S=2048 D=1024 H=16 DC=64 HC=16 HD=64
// ---------------------------------------------------------------------------

#define B_  2
#define S_  2048
#define D_  1024
#define H_  16
#define DC_ 64
#define HC_ 16
#define HD_ 64
#define HN_ (H_ * HC_)   // 256

typedef __attribute__((ext_vector_type(16))) _Float16 v16h;
typedef __attribute__((ext_vector_type(8)))  _Float16 v8h;
typedef __attribute__((ext_vector_type(8)))  float    v8f;
typedef __attribute__((ext_vector_type(2)))  float    v2f;

// ---------------------------------------------------------------------------
// f32 -> f16 convert
// ---------------------------------------------------------------------------
__global__ void cvt_f16_kernel(const float* __restrict__ src,
                               _Float16* __restrict__ dst, int n) {
  int i = blockIdx.x * blockDim.x + threadIdx.x;
  if (i < n) dst[i] = (_Float16)src[i];
}

// ---------------------------------------------------------------------------
// W[K,N] f32 -> WT[N,K] f16 (transpose + convert; coalesced reads)
// ---------------------------------------------------------------------------
__global__ void transpose_cvt_kernel(const float* __restrict__ W,
                                     _Float16* __restrict__ WT,
                                     int K, int N) {
  int i = blockIdx.x * blockDim.x + threadIdx.x;
  if (i >= K * N) return;
  int k = i / N, n = i % N;
  WT[(size_t)n * K + k] = (_Float16)W[i];
}

// ---------------------------------------------------------------------------
// z[b,h,s,c] = coords[b,s,:] @ Wc[:, h*HC+c] + bc   (DC=64 inner)
// ---------------------------------------------------------------------------
__global__ void coord_proj_kernel(const float* __restrict__ coords,
                                  const float* __restrict__ Wc,
                                  const float* __restrict__ bc,
                                  float* __restrict__ z) {
  int idx = blockIdx.x * blockDim.x + threadIdx.x;      // over B*H*S*HC
  if (idx >= B_ * H_ * S_ * HC_) return;
  int c = idx % HC_;
  int s = (idx / HC_) % S_;
  int h = (idx / (HC_ * S_)) % H_;
  int b = idx / (HC_ * S_ * H_);
  int n = h * HC_ + c;
  const float* crow = coords + ((size_t)b * S_ + s) * DC_;
  float acc = bc[n];
  #pragma unroll 8
  for (int d = 0; d < DC_; ++d) acc += crow[d] * Wc[(size_t)d * HN_ + n];
  z[idx] = acc;
}

// ---------------------------------------------------------------------------
// zsq[b,h,s] = sum_c z[b,h,s,c]^2
// ---------------------------------------------------------------------------
__global__ void zsq_kernel(const float* __restrict__ z,
                           float* __restrict__ zsq) {
  int idx = blockIdx.x * blockDim.x + threadIdx.x;      // over B*H*S
  if (idx >= B_ * H_ * S_) return;
  const float* zr = z + (size_t)idx * HC_;
  float acc = 0.f;
  #pragma unroll
  for (int c = 0; c < HC_; ++c) acc += zr[c] * zr[c];
  zsq[idx] = acc;
}

// ---------------------------------------------------------------------------
// updated_coords = coords @ Wn + bn   (tiny: 64x64)
// ---------------------------------------------------------------------------
__global__ void coords_out_kernel(const float* __restrict__ coords,
                                  const float* __restrict__ Wn,
                                  const float* __restrict__ bn,
                                  float* __restrict__ out) {
  int idx = blockIdx.x * blockDim.x + threadIdx.x;      // over B*S*DC
  if (idx >= B_ * S_ * DC_) return;
  int n  = idx % DC_;
  int bs = idx / DC_;
  const float* crow = coords + (size_t)bs * DC_;
  float acc = bn[n];
  #pragma unroll 8
  for (int d = 0; d < DC_; ++d) acc += crow[d] * Wn[(size_t)d * DC_ + n];
  out[idx] = acc;
}

// ---------------------------------------------------------------------------
// f16 WMMA GEMM: C[M,N]f32 = A[M,K]f16 @ WT[N,K]f16^T + bias
//   block = 128 threads = 4 waves; block tile = 16(M) x 64(N); 16x16 per wave.
//   mode 0: f32 store row-major [M,N] into Cf  (out projection -> d_out)
//   mode 1: f16 store transposed-per-head value: Ch[b,h,hd,s]
// ---------------------------------------------------------------------------
__global__ __launch_bounds__(128)
void gemm_wmma_f16_kernel(const _Float16* __restrict__ A,
                          const _Float16* __restrict__ WT,
                          const float* __restrict__ bias,
                          float* __restrict__ Cf,
                          _Float16* __restrict__ Ch,
                          int M, int N, int K, int mode) {
  const int lane = threadIdx.x & 31;
  const int wave = threadIdx.x >> 5;        // 0..3
  const int half = lane >> 4;               // 0/1: K-half selector
  const int lidx = lane & 15;
  const int m0 = blockIdx.x * 16;
  const int n0 = blockIdx.y * 64 + wave * 16;

  // A fragment rows: row = lidx; half selects 16 contiguous K-values (+0 / +16)
  const _Float16* arow = A  + (size_t)(m0 + lidx) * K + half * 16;
  // B fragment: 16 contiguous K-values of column (n0+lidx); half selects +16
  const _Float16* brow = WT + (size_t)(n0 + lidx) * K + half * 16;

  v8f acc = {};
  for (int k = 0; k < K; k += 32) {
    v8h alo = *(const v8h*)(arow + k);        // K: k+half*16  .. +7
    v8h ahi = *(const v8h*)(arow + k + 8);    // K: k+half*16+8.. +7
    v16h a, b;
    #pragma unroll
    for (int i = 0; i < 8; ++i) { a[i] = alo[i]; a[8 + i] = ahi[i]; }
    b = *(const v16h*)(brow + k);             // K: k+half*16 .. +15
    acc = __builtin_amdgcn_wmma_f32_16x16x32_f16(
        false, a, false, b, (short)0, acc, false, false);
  }

  // C layout: VGPR j -> row m0 + j + 8*half, col n0 + lidx
  #pragma unroll
  for (int j = 0; j < 8; ++j) {
    int m = m0 + j + 8 * half;
    int n = n0 + lidx;
    float v = acc[j] + bias[n];
    if (mode == 0) {
      Cf[(size_t)m * N + n] = v;
    } else {
      int b  = m / S_, s  = m % S_;
      int h  = n / HD_, hd = n % HD_;
      // VT[b,h,hd,s] f16 (contiguous in s for attention B-fragments)
      Ch[(((size_t)b * H_ + h) * HD_ + hd) * S_ + s] = (_Float16)v;
    }
  }
}

// ---------------------------------------------------------------------------
// Fused gravity attention (flash-style online softmax), one wave per
// (b,h, 16-query tile).  32 keys per iteration:
//   * z key tile (32x16 f32, 2KB contiguous) staged via
//     GLOBAL_LOAD_ASYNC_TO_LDS_B128 (ASYNCcnt), drained with s_wait_asynccnt
//   * scores  : V_WMMA_F32_16X16X4_F32 (f32 precision into softmax)
//   * P @ V   : V_WMMA_F32_16X16X32_F16 (P -> f16 through LDS, V pre-f16)
//   out f16 -> attn_out[b, s, h*HD+hd]  (row-major [B*S, D] for out-proj GEMM)
// ---------------------------------------------------------------------------
__global__ __launch_bounds__(32)
void gravity_attn_kernel(const float* __restrict__ z,      // [B,H,S,HC] f32
                         const float* __restrict__ zsq,    // [B,H,S]    f32
                         const _Float16* __restrict__ VT,  // [B,H,HD,S] f16
                         const float* __restrict__ gamma,  // [H]
                         _Float16* __restrict__ attn_out)  // [B,S,H*HD] f16
{
  __shared__ float    zkLds[32 * 16];     // staged z key tile (2KB)
  __shared__ _Float16 pLds[16 * 48];      // P tile f16, row stride 48

  const int bh   = blockIdx.x;            // b*H + h
  const int h    = bh % H_;
  const int b    = bh / H_;
  const int q0   = blockIdx.y * 16;
  const int lane = threadIdx.x & 31;
  const int half = lane >> 4;
  const int lidx = lane & 15;

  const float gx = gamma[h];
  const float g  = fmaxf(gx, 0.f) + log1pf(expf(-fabsf(gx)));  // softplus

  const float*    zbh   = z   + (size_t)bh * S_ * HC_;
  const float*    zsqbh = zsq + (size_t)bh * S_;
  const _Float16* vtbh  = VT  + (size_t)bh * HD_ * S_;

  // LDS byte offsets (generic LDS pointers carry the offset in the low bits)
  const unsigned zk_base = (unsigned)(size_t)&zkLds[0];

  // A fragments of the query tile (reused for every key tile):
  // f32 16x16x4 A layout: lane -> row lidx, K pair (k0 + 2*half, +1)
  v2f afr[4];
  #pragma unroll
  for (int kk = 0; kk < 4; ++kk) {
    const float* p = zbh + (size_t)(q0 + lidx) * HC_ + kk * 4 + 2 * half;
    afr[kk].x = p[0]; afr[kk].y = p[1];
  }
  float zsq_q[8];
  #pragma unroll
  for (int j = 0; j < 8; ++j) zsq_q[j] = zsqbh[q0 + j + 8 * half];

  v8f acc[4]; float mrow[8], lrow[8];
  #pragma unroll
  for (int nc = 0; nc < 4; ++nc) acc[nc] = (v8f){};
  #pragma unroll
  for (int j = 0; j < 8; ++j) { mrow[j] = -3.0e38f; lrow[j] = 0.f; }

  for (int t0 = 0; t0 < S_; t0 += 32) {
    // ---- async-stage z[t0..t0+31][0..15] (contiguous 2KB) into LDS ----
    {
      unsigned long long gaddr =
          (unsigned long long)(size_t)(zbh + (size_t)t0 * HC_) + lane * 16ull;
      unsigned laddr = zk_base + lane * 16u;
      asm volatile("s_wait_dscnt 0" ::: "memory");  // WAR vs prior LDS reads
      asm volatile(
          "global_load_async_to_lds_b128 %0, %1, off\n\t"
          "global_load_async_to_lds_b128 %0, %1, off offset:512\n\t"
          "global_load_async_to_lds_b128 %0, %1, off offset:1024\n\t"
          "global_load_async_to_lds_b128 %0, %1, off offset:1536"
          :: "v"(laddr), "v"(gaddr) : "memory");
    }
    // prefetch next V tile region while the async copy runs
    if (t0 + 32 < S_)
      __builtin_prefetch(vtbh + (size_t)lane * S_ + t0 + 32, 0, 3);
    asm volatile("s_wait_asynccnt 0" ::: "memory");

    // ---- scores for the two 16-key subtiles (f32 WMMA, K=HC=16) ----
    v8f sc[2];
    #pragma unroll
    for (int st = 0; st < 2; ++st) {
      sc[st] = (v8f){};
      #pragma unroll
      for (int kk = 0; kk < 4; ++kk) {
        const float* p = &zkLds[(st * 16 + lidx) * 16 + kk * 4 + 2 * half];
        v2f bfr; bfr.x = p[0]; bfr.y = p[1];
        sc[st] = __builtin_amdgcn_wmma_f32_16x16x4_f32(
            false, afr[kk], false, bfr, (short)0, sc[st], false, false);
      }
    }
    const float zsq_t0 = zsqbh[t0 + lidx];
    const float zsq_t1 = zsqbh[t0 + 16 + lidx];

    // ---- online softmax over the 32-key block ----
    #pragma unroll
    for (int j = 0; j < 8; ++j) {
      float sq0 = fmaxf(zsq_q[j] + zsq_t0 - 2.0f * sc[0][j], 0.f);
      float sq1 = fmaxf(zsq_q[j] + zsq_t1 - 2.0f * sc[1][j], 0.f);
      float s0 = -g * sq0, s1 = -g * sq1;
      float rmax = fmaxf(s0, s1);
      #pragma unroll
      for (int off = 1; off < 16; off <<= 1)
        rmax = fmaxf(rmax, __shfl_xor(rmax, off, 32));
      float mnew  = fmaxf(mrow[j], rmax);
      float scale = expf(mrow[j] - mnew);
      float p0 = expf(s0 - mnew);
      float p1 = expf(s1 - mnew);
      float rsum = p0 + p1;
      #pragma unroll
      for (int off = 1; off < 16; off <<= 1)
        rsum += __shfl_xor(rsum, off, 32);
      lrow[j] = lrow[j] * scale + rsum;
      mrow[j] = mnew;
      #pragma unroll
      for (int nc = 0; nc < 4; ++nc) acc[nc][j] *= scale;
      const int row = j + 8 * half;                       // C-layout row
      pLds[row * 48 + lidx]      = (_Float16)p0;
      pLds[row * 48 + 16 + lidx] = (_Float16)p1;
    }
    __syncthreads();

    // ---- acc += P(16x32,f16) @ V(32x16 per hd-chunk, f16) via f16 WMMA ----
    v16h ap;
    {
      v8h alo = *(const v8h*)&pLds[lidx * 48 + half * 16];       // K half*16 .. +7
      v8h ahi = *(const v8h*)&pLds[lidx * 48 + 8 + half * 16];   // K half*16+8 .. +15
      #pragma unroll
      for (int i = 0; i < 8; ++i) { ap[i] = alo[i]; ap[8 + i] = ahi[i]; }
    }
    #pragma unroll
    for (int nc = 0; nc < 4; ++nc) {
      // B[k][n] = V[t0+k][nc*16+n] = VT[nc*16+n][t0+k]; lane col n = lidx
      const _Float16* bp = vtbh + (size_t)(nc * 16 + lidx) * S_ + t0 + half * 16;
      v16h bf = *(const v16h*)bp;
      acc[nc] = __builtin_amdgcn_wmma_f32_16x16x32_f16(
          false, ap, false, bf, (short)0, acc[nc], false, false);
    }
    __syncthreads();
  }

  // ---- finalize: out = acc / l, store f16 into [b, s, h*HD + hd] ----
  #pragma unroll
  for (int j = 0; j < 8; ++j) {
    const int srow = q0 + j + 8 * half;
    const float inv = 1.0f / lrow[j];
    _Float16* op = attn_out + ((size_t)b * S_ + srow) * (size_t)D_ + h * HD_;
    #pragma unroll
    for (int nc = 0; nc < 4; ++nc)
      op[nc * 16 + lidx] = (_Float16)(acc[nc][j] * inv);
  }
}

// ---------------------------------------------------------------------------
// Launcher
// ---------------------------------------------------------------------------
extern "C" void kernel_launch(void* const* d_in, const int* in_sizes, int n_in,
                              void* d_out, int out_size, void* d_ws, size_t ws_size,
                              hipStream_t stream) {
  const float* hidden = (const float*)d_in[0];
  const float* coords = (const float*)d_in[1];
  const float* Wv     = (const float*)d_in[2];
  const float* bv     = (const float*)d_in[3];
  const float* Wc     = (const float*)d_in[4];
  const float* bc     = (const float*)d_in[5];
  const float* Wn     = (const float*)d_in[6];
  const float* bn     = (const float*)d_in[7];
  const float* Wo     = (const float*)d_in[8];
  const float* bo     = (const float*)d_in[9];
  const float* gamma  = (const float*)d_in[10];

  float* out_hidden = (float*)d_out;                       // [B,S,D]
  float* out_coords = out_hidden + (size_t)B_ * S_ * D_;   // [B,S,DC]

  // ---- workspace carve-up ----
  char* p = (char*)d_ws;
  auto carve = [&](size_t bytes) {
    char* r = p;
    p += (bytes + 255) & ~(size_t)255;
    return r;
  };
  _Float16* hid16  = (_Float16*)carve((size_t)B_ * S_ * D_ * 2);  // hidden f16
  _Float16* WvT    = (_Float16*)carve((size_t)D_ * D_ * 2);       // Wv^T f16
  _Float16* WoT    = (_Float16*)carve((size_t)D_ * D_ * 2);       // Wo^T f16
  _Float16* VT     = (_Float16*)carve((size_t)B_ * S_ * D_ * 2);  // V^T [b,h,hd,s] f16
  float*    z      = (float*)   carve((size_t)B_ * H_ * S_ * HC_ * 4);
  float*    zsq    = (float*)   carve((size_t)B_ * H_ * S_ * 4);
  _Float16* aout16 = (_Float16*)carve((size_t)B_ * S_ * D_ * 2);  // attn out f16
  (void)ws_size;

  const int nBSD = B_ * S_ * D_;

  // 1) converts / transposes
  cvt_f16_kernel<<<(nBSD + 255) / 256, 256, 0, stream>>>(hidden, hid16, nBSD);
  transpose_cvt_kernel<<<(D_ * D_ + 255) / 256, 256, 0, stream>>>(Wv, WvT, D_, D_);
  transpose_cvt_kernel<<<(D_ * D_ + 255) / 256, 256, 0, stream>>>(Wo, WoT, D_, D_);

  // 2) coordinate pipeline
  {
    int n = B_ * H_ * S_ * HC_;
    coord_proj_kernel<<<(n + 255) / 256, 256, 0, stream>>>(coords, Wc, bc, z);
  }
  {
    int n = B_ * H_ * S_;
    zsq_kernel<<<(n + 255) / 256, 256, 0, stream>>>(z, zsq);
  }
  {
    int n = B_ * S_ * DC_;
    coords_out_kernel<<<(n + 255) / 256, 256, 0, stream>>>(coords, Wn, bn, out_coords);
  }

  // 3) value projection: V = hidden @ Wv + bv  -> VT[b,h,hd,s] f16
  {
    dim3 grid((B_ * S_) / 16, D_ / 64);
    gemm_wmma_f16_kernel<<<grid, 128, 0, stream>>>(
        hid16, WvT, bv, nullptr, VT, B_ * S_, D_, D_, /*mode=*/1);
  }

  // 4) fused gravity attention -> attn_out f16 [B,S,D]
  {
    dim3 grid(B_ * H_, S_ / 16);
    gravity_attn_kernel<<<grid, 32, 0, stream>>>(z, zsq, VT, gamma, aout16);
  }

  // 5) output projection: updated_hidden = attn_out @ Wo + bo
  {
    dim3 grid((B_ * S_) / 16, D_ / 64);
    gemm_wmma_f16_kernel<<<grid, 128, 0, stream>>>(
        aout16, WoT, bo, out_hidden, nullptr, B_ * S_, D_, D_, /*mode=*/0);
  }
}
